// SeqWeightAttention_65377992179725
// MI455X (gfx1250) — matrix operand order repaired and checked by reading, not compiled, not run on hardware
//
#include <hip/hip_runtime.h>
#include <hip/hip_bf16.h>
#include <math.h>

#define RR 512
#define CC 768
#define DD 256
#define HH 8
#define HD 32
#define NEG_MIN (-3.4028234663852886e38f)
#define SCALE 0.17677669529663687f  // 1/sqrt(32)

typedef __attribute__((ext_vector_type(16))) __bf16 v16bf;
typedef __attribute__((ext_vector_type(2)))  __bf16 bf16x2;
typedef __attribute__((ext_vector_type(8)))  float  v8f;

union V16 {
    v16bf    v;
    unsigned u[8];
    uint4    q[2];
};

// Single v_cvt_pk_bf16_f32: build a native 2-elt __bf16 vector, bitcast to u32.
static __device__ __forceinline__ unsigned pack2bf(float x, float y) {
    bf16x2 p;
    p.x = (__bf16)x;
    p.y = (__bf16)y;
    return __builtin_bit_cast(unsigned, p);
}

// -------- Kernel A: Wqk[c,h,d] = sum_hd (scale * (query[c]·Wq[h*32+hd])) * Wk[h*32+hd,d]
// grid = 768 (c), block = 256 (d / o)
__global__ void __launch_bounds__(256) wqk_fuse_kernel(
    const float* __restrict__ MSA, const float* __restrict__ Wq,
    const float* __restrict__ Wk, unsigned short* __restrict__ Wqk) {
    const int c = blockIdx.x;
    const int t = threadIdx.x;  // 0..255
    __shared__ float Qs[DD];

    // query[c,d] = MSA[r=0, c, d]
    const float* qrow  = MSA + (size_t)c * DD;
    const float* wqrow = Wq + (size_t)t * DD;
    float q = 0.f;
    for (int d = 0; d < DD; ++d) q = fmaf(qrow[d], wqrow[d], q);
    Qs[t] = q * SCALE;
    __syncthreads();

    for (int h = 0; h < HH; ++h) {
        float s = 0.f;
        #pragma unroll 8
        for (int j = 0; j < HD; ++j)
            s = fmaf(Qs[h * HD + j], Wk[(size_t)(h * HD + j) * DD + t], s);
        Wqk[(size_t)(c * HH + h) * DD + t] =
            __builtin_bit_cast(unsigned short, (__bf16)s);
    }
}

// -------- Kernel B: per-c score GEMM (512x8x256) via v_wmma_f32_16x16x32_bf16,
// masked softmax over r, transposed write-out.
// grid = 768 (c), block = 128 (4 waves)
__global__ void __launch_bounds__(128) attn_kernel(
    const float* __restrict__ MSA, const int* __restrict__ seq_mask,
    const unsigned short* __restrict__ Wqk, float* __restrict__ out) {
    const int c    = blockIdx.x;
    const int tid  = threadIdx.x;
    const int lane = tid & 31;
    const int wave = tid >> 5;   // 0..3
    const int half = lane >> 4;  // 0/1
    const int l15  = lane & 15;

    __shared__ float S[HH * RR];  // scores [h][r], later exp'd scores
    __shared__ int   smask[RR];
    __shared__ float red[128];
    __shared__ float mx[HH], inv[HH];

    for (int r0 = tid; r0 < RR; r0 += 128) smask[r0] = seq_mask[r0];
    __syncthreads();

    // ---- B matrix: Wqk row for this c; N = head (zero-pad n>=8). Loaded once.
    // B layout (32x16 bf16): lane n<16 = column N; lanes 0-15 hold K=0..15,
    // lanes 16-31 hold K=16..31, two K per VGPR.
    const int n = l15;
    V16 b[8];
    if (n < HH) {
        const uint4* bp = (const uint4*)(Wqk + (size_t)(c * HH + n) * DD);
        #pragma unroll
        for (int kc = 0; kc < 8; ++kc) {
            const int e = (kc * 32 + half * 16) >> 3;  // uint4 index (8 ushort each)
            b[kc].q[0] = bp[e];
            b[kc].q[1] = bp[e + 1];
        }
    } else {
        #pragma unroll
        for (int kc = 0; kc < 8; ++kc) {
            b[kc].q[0] = make_uint4(0u, 0u, 0u, 0u);
            b[kc].q[1] = make_uint4(0u, 0u, 0u, 0u);
        }
    }

    // ---- 8 M-tiles of 16 rows per wave; K = 256 in 8 chunks of 32
    for (int tile = 0; tile < 8; ++tile) {
        const int   t16  = (wave * 8 + tile) * 16;
        const int   r    = t16 + l15;  // A: lanes 0-15 and 16-31 both hold M=0..15
        const float* arow = MSA + ((size_t)r * CC + c) * DD;
        if (tile < 7) __builtin_prefetch(arow + (size_t)16 * CC * DD, 0, 1);

        v8f acc = {0.f, 0.f, 0.f, 0.f, 0.f, 0.f, 0.f, 0.f};
        #pragma unroll
        for (int kc = 0; kc < 8; ++kc) {
            // A layout (16x32 bf16): half0 K=0..15, half1 K=16..31 (contiguous)
            V16 a;
            #pragma unroll
            for (int p = 0; p < 4; ++p) {
                const int ks = kc * 32 + half * 16 + p * 4;
                const float4 f = *(const float4*)(arow + ks);
                a.u[2 * p]     = pack2bf(f.x, f.y);
                a.u[2 * p + 1] = pack2bf(f.z, f.w);
            }
            acc = __builtin_amdgcn_wmma_f32_16x16x32_bf16(
                false, a.v, false, b[kc].v, (short)0, acc, false, false);
        }
        // D layout: lane n = N (head); VGPR v -> M = v + half*8
        if (n < HH) {
            #pragma unroll
            for (int v = 0; v < 8; ++v) {
                const int rr = t16 + v + half * 8;
                S[n * RR + rr] = (smask[rr] == 0) ? NEG_MIN : acc[v];
            }
        }
    }
    __syncthreads();

    // ---- softmax over r for each head: 16 threads per head
    const int h = tid >> 4;  // 0..7
    const int j = tid & 15;
    float m = NEG_MIN;
    for (int r0 = j; r0 < RR; r0 += 16) m = fmaxf(m, S[h * RR + r0]);
    red[tid] = m;
    __syncthreads();
    if (j == 0) {
        float mm = red[tid];
        for (int k = 1; k < 16; ++k) mm = fmaxf(mm, red[tid + k]);
        mx[h] = mm;
    }
    __syncthreads();
    const float mh = mx[h];
    float s = 0.f;
    for (int r0 = j; r0 < RR; r0 += 16) {
        const float e = __expf(S[h * RR + r0] - mh);
        S[h * RR + r0] = e;  // cache exp'd value for the write pass
        s += e;
    }
    red[tid] = s;
    __syncthreads();
    if (j == 0) {
        float ss = 0.f;
        for (int k = 0; k < 16; ++k) ss += red[tid + k];
        inv[h] = 1.0f / ss;
    }
    __syncthreads();

    // ---- transposed write: out[((r*768)+c)*8 + h]; 32B contiguous per (r,c)
    #pragma unroll
    for (int qq = 0; qq < 4; ++qq) {
        const int r0 = tid * 4 + qq;
        float p[8];
        #pragma unroll
        for (int hh = 0; hh < HH; ++hh)
            p[hh] = S[hh * RR + r0] * inv[hh];
        float4* po = (float4*)(out + ((size_t)r0 * CC + c) * HH);
        po[0] = make_float4(p[0], p[1], p[2], p[3]);
        po[1] = make_float4(p[4], p[5], p[6], p[7]);
    }
}

extern "C" void kernel_launch(void* const* d_in, const int* in_sizes, int n_in,
                              void* d_out, int out_size, void* d_ws, size_t ws_size,
                              hipStream_t stream) {
    const float* MSA      = (const float*)d_in[0];
    const int*   seq_mask = (const int*)d_in[1];
    const float* Wq       = (const float*)d_in[2];
    const float* Wk       = (const float*)d_in[3];
    float*       out      = (float*)d_out;
    // workspace: Wqk bf16, 768*8*256*2 = 3,145,728 bytes
    unsigned short* Wqk = (unsigned short*)d_ws;

    wqk_fuse_kernel<<<CC, 256, 0, stream>>>(MSA, Wq, Wk, Wqk);
    attn_kernel<<<CC, 128, 0, stream>>>(MSA, seq_mask, Wqk, out);
}
